// SwapScoringHead_88940182765945
// MI455X (gfx1250) — matrix operand order repaired and checked by reading, not compiled, not run on hardware
//
#include <hip/hip_runtime.h>
#include <cmath>

// ---------------------------------------------------------------------------
// Types for CDNA5 WMMA (wave32): v_wmma_f32_16x16x32_bf16
// ---------------------------------------------------------------------------
typedef __bf16 bf16;
typedef __attribute__((ext_vector_type(16))) __bf16 v16bf;
typedef __attribute__((ext_vector_type(8)))  __bf16 v8bf;
typedef __attribute__((ext_vector_type(8)))  float  v8f;
typedef __attribute__((ext_vector_type(4)))  float  v4f;

__device__ __forceinline__ float gelu_exact(float x) {
    // jax.nn.gelu(approximate=False): 0.5*x*(1+erf(x/sqrt(2)))
    return 0.5f * x * (1.0f + erff(x * 0.70710678118654752440f));
}
__device__ __forceinline__ float softplus_f(float x) {
    // numerically stable log(1+exp(x))
    return fmaxf(x, 0.0f) + log1pf(expf(-fabsf(x)));
}

// Problem constants
#define BB   8
#define NN   2048
#define DD   512
#define SS   4096
#define MTOT (BB * SS)      // 32768 rows
#define K1   (4 * DD)       // 2048
#define N1   DD             // 512
#define K2   DD             // 512
#define N2   (DD / 2)       // 256

// ---------------------------------------------------------------------------
// Kernel 0: transpose-convert weights to bf16, K-contiguous per output column
//   W1 (2048,512) f32  -> W1t (512,2048) bf16
//   W2 (512,256)  f32  -> W2t (256,512)  bf16
// ---------------------------------------------------------------------------
__global__ void prep_weights(const float* __restrict__ W1,
                             const float* __restrict__ W2,
                             bf16* __restrict__ W1t,
                             bf16* __restrict__ W2t) {
    int i = blockIdx.x * blockDim.x + threadIdx.x;
    if (i < K1 * N1) {
        int k = i >> 9;          // /512
        int n = i & (N1 - 1);
        W1t[(size_t)n * K1 + k] = (bf16)W1[i];
    }
    if (i < K2 * N2) {
        int k = i >> 8;          // /256
        int n = i & (N2 - 1);
        W2t[(size_t)n * K2 + k] = (bf16)W2[i];
    }
}

// ---------------------------------------------------------------------------
// Kernel 1: fused gather + GEMM1 (M=32768, K=2048, N=512) + bias + exact GELU
// Block: 256 thr = 8 waves arranged 2(M) x 4(N); block tile 128 x 256.
// Wave tile 64x64 = 4x4 WMMA subtiles (128 accumulator VGPRs).
// A fragments gathered from h on the fly (f32 -> bf16 in registers).
// ---------------------------------------------------------------------------
__launch_bounds__(256, 1)
__global__ void gemm1_gather_gelu(const float* __restrict__ H,
                                  const int*   __restrict__ IDX,
                                  const bf16*  __restrict__ W1t,
                                  const float* __restrict__ b1,
                                  bf16* __restrict__ X1) {
    const int lane  = threadIdx.x & 31;
    const int wave  = threadIdx.x >> 5;
    const int waveM = wave >> 2;                 // 0..1
    const int waveN = wave & 3;                  // 0..3
    const int Mbase = blockIdx.y * 128 + waveM * 64;
    const int Nbase = blockIdx.x * 256 + waveN * 64;
    const int ln    = lane & 15;
    const int hf    = lane >> 4;                 // half-wave select

    v8f acc[4][4];
#pragma unroll
    for (int mi = 0; mi < 4; ++mi)
#pragma unroll
        for (int ni = 0; ni < 4; ++ni)
#pragma unroll
            for (int e = 0; e < 8; ++e) acc[mi][ni][e] = 0.0f;

    for (int kb = 0; kb < K1; kb += 32) {
        // --- B fragments (32x16): lane = col n; elems 0-7 @ K=kb+8h.., 8-15 @ K=kb+16+8h.. ---
        v16bf bfr[4];
#pragma unroll
        for (int ni = 0; ni < 4; ++ni) {
            const int n = Nbase + ni * 16 + ln;
            const bf16* p = W1t + (size_t)n * K1 + kb + hf * 8;
            v8bf lo = *(const v8bf*)p;
            v8bf up = *(const v8bf*)(p + 16);
            v16bf t;
#pragma unroll
            for (int e = 0; e < 8; ++e) { t[e] = lo[e]; t[e + 8] = up[e]; }
            bfr[ni] = t;
        }
        // --- A fragments (16x32) gathered: elems 0-7 @ K=k0.., 8-15 @ K=k0+16.. ---
        const int k0 = kb + hf * 8;
        const int j  = k0 >> 9;                  // which of 4 gathered nodes
        const int d0 = k0 & (DD - 1);            // offset inside node row
#pragma unroll
        for (int mi = 0; mi < 4; ++mi) {
            const int m    = Mbase + mi * 16 + ln;          // row = b*S + s
            const int node = IDX[m * 4 + j];
            const float* p = H + ((size_t)(m >> 12) * NN + node) * DD + d0;
            v4f a0 = *(const v4f*)(p);
            v4f a1 = *(const v4f*)(p + 4);
            v4f a2 = *(const v4f*)(p + 16);
            v4f a3 = *(const v4f*)(p + 20);
            v16bf afr;
#pragma unroll
            for (int e = 0; e < 4; ++e) {
                afr[e]      = (bf16)a0[e];
                afr[e + 4]  = (bf16)a1[e];
                afr[e + 8]  = (bf16)a2[e];
                afr[e + 12] = (bf16)a3[e];
            }
#pragma unroll
            for (int ni = 0; ni < 4; ++ni)
                acc[mi][ni] = __builtin_amdgcn_wmma_f32_16x16x32_bf16(
                    false, afr, false, bfr[ni], (short)0, acc[mi][ni], false, false);
        }
    }

    // Epilogue: + b1, exact GELU, store bf16 activations
#pragma unroll
    for (int mi = 0; mi < 4; ++mi) {
        const int mrow = Mbase + mi * 16 + hf * 8;
#pragma unroll
        for (int ni = 0; ni < 4; ++ni) {
            const int n = Nbase + ni * 16 + ln;
            const float bias = b1[n];
#pragma unroll
            for (int r = 0; r < 8; ++r) {
                float g = gelu_exact(acc[mi][ni][r] + bias);
                X1[(size_t)(mrow + r) * N1 + n] = (bf16)g;
            }
        }
    }
}

// ---------------------------------------------------------------------------
// Kernel 2: GEMM2 (M=32768, K=512, N=256) + bias + GELU, fused W3 dot,
// + b3 + softplus. Block tile 128 x 256 (full N), 8 waves 2(M)x4(N).
// ---------------------------------------------------------------------------
__launch_bounds__(256, 1)
__global__ void gemm2_head(const bf16*  __restrict__ X1,
                           const bf16*  __restrict__ W2t,
                           const float* __restrict__ b2,
                           const float* __restrict__ W3,
                           const float* __restrict__ b3,
                           float* __restrict__ out) {
    __shared__ float part[128][4];

    const int lane  = threadIdx.x & 31;
    const int wave  = threadIdx.x >> 5;
    const int waveM = wave >> 2;
    const int waveN = wave & 3;
    const int MbaseBlk = blockIdx.x * 128;
    const int Mbase = MbaseBlk + waveM * 64;
    const int Nbase = waveN * 64;
    const int ln    = lane & 15;
    const int hf    = lane >> 4;

    v8f acc[4][4];
#pragma unroll
    for (int mi = 0; mi < 4; ++mi)
#pragma unroll
        for (int ni = 0; ni < 4; ++ni)
#pragma unroll
            for (int e = 0; e < 8; ++e) acc[mi][ni][e] = 0.0f;

    for (int kb = 0; kb < K2; kb += 32) {
        v16bf bfr[4];
#pragma unroll
        for (int ni = 0; ni < 4; ++ni) {
            const int n = Nbase + ni * 16 + ln;
            const bf16* p = W2t + (size_t)n * K2 + kb + hf * 8;
            v8bf lo = *(const v8bf*)p;
            v8bf up = *(const v8bf*)(p + 16);
            v16bf t;
#pragma unroll
            for (int e = 0; e < 8; ++e) { t[e] = lo[e]; t[e + 8] = up[e]; }
            bfr[ni] = t;
        }
        const int k0 = kb + hf * 8;
#pragma unroll
        for (int mi = 0; mi < 4; ++mi) {
            const int m = Mbase + mi * 16 + ln;
            const bf16* p = X1 + (size_t)m * K2 + k0;
            v8bf lo = *(const v8bf*)p;          // K = k0 .. k0+7
            v8bf up = *(const v8bf*)(p + 16);   // K = k0+16 .. k0+23
            v16bf afr;
#pragma unroll
            for (int e = 0; e < 8; ++e) { afr[e] = lo[e]; afr[e + 8] = up[e]; }
#pragma unroll
            for (int ni = 0; ni < 4; ++ni)
                acc[mi][ni] = __builtin_amdgcn_wmma_f32_16x16x32_bf16(
                    false, afr, false, bfr[ni], (short)0, acc[mi][ni], false, false);
        }
    }

    // Head: gelu(acc + b2) * W3, reduce over N (16 lanes per half, 4 N-subtiles,
    // 4 N-waves), then +b3 and softplus.
#pragma unroll
    for (int mi = 0; mi < 4; ++mi) {
        float t[8];
#pragma unroll
        for (int r = 0; r < 8; ++r) t[r] = 0.0f;
#pragma unroll
        for (int ni = 0; ni < 4; ++ni) {
            const int n = Nbase + ni * 16 + ln;
            const float bias = b2[n];
            const float w3v  = W3[n];
#pragma unroll
            for (int r = 0; r < 8; ++r)
                t[r] += gelu_exact(acc[mi][ni][r] + bias) * w3v;
        }
#pragma unroll
        for (int r = 0; r < 8; ++r) {
            float v = t[r];
            v += __shfl_xor(v, 1, 32);
            v += __shfl_xor(v, 2, 32);
            v += __shfl_xor(v, 4, 32);
            v += __shfl_xor(v, 8, 32);          // stays within 16-lane half
            if (ln == 0) {
                const int lm = (Mbase - MbaseBlk) + mi * 16 + hf * 8 + r; // 0..127
                part[lm][waveN] = v;
            }
        }
    }
    __syncthreads();

    const int tid = threadIdx.x;
    if (tid < 128) {
        float s = part[tid][0] + part[tid][1] + part[tid][2] + part[tid][3] + b3[0];
        out[MbaseBlk + tid] = softplus_f(s);
    }
}

// ---------------------------------------------------------------------------
// Host-side launcher
// ---------------------------------------------------------------------------
extern "C" void kernel_launch(void* const* d_in, const int* in_sizes, int n_in,
                              void* d_out, int out_size, void* d_ws, size_t ws_size,
                              hipStream_t stream) {
    const float* H   = (const float*)d_in[0];
    const int*   IDX = (const int*)d_in[1];
    const float* W1  = (const float*)d_in[2];
    const float* b1  = (const float*)d_in[3];
    const float* W2  = (const float*)d_in[4];
    const float* b2  = (const float*)d_in[5];
    const float* W3  = (const float*)d_in[6];
    const float* b3  = (const float*)d_in[7];
    float* out = (float*)d_out;

    char* ws = (char*)d_ws;
    bf16* W1t = (bf16*)(ws);
    bf16* W2t = (bf16*)(ws + (size_t)K1 * N1 * sizeof(bf16));
    bf16* X1  = (bf16*)(ws + (size_t)K1 * N1 * sizeof(bf16)
                           + (size_t)K2 * N2 * sizeof(bf16));

    // 1) weights -> bf16, transposed for K-contiguous B fragments
    prep_weights<<<(K1 * N1 + 255) / 256, 256, 0, stream>>>(W1, W2, W1t, W2t);

    // 2) gather + GEMM1 + GELU  (grid: N-tiles x M-tiles)
    gemm1_gather_gelu<<<dim3(N1 / 256, MTOT / 128), 256, 0, stream>>>(
        H, IDX, W1t, b1, X1);

    // 3) GEMM2 + GELU + W3 dot + softplus
    gemm2_head<<<MTOT / 128, 256, 0, stream>>>(X1, W2t, b2, W3, b3, out);
}
